// AdaptiveSampler_30227979829999
// MI455X (gfx1250) — matrix operand order repaired and checked by reading, not compiled, not run on hardware
//
#include <hip/hip_runtime.h>
#include <math.h>

typedef __attribute__((ext_vector_type(16))) __bf16 v16bf;
typedef __attribute__((ext_vector_type(4)))  __bf16 v4bf;
typedef __attribute__((ext_vector_type(8)))  float  v8f;

#define COS_EPS   1e-6f
#define FX_SCALE  1099511627776.0      /* 2^40 */
#define FX_INV    (1.0 / 1099511627776.0)

// order-monotonic encoding of float for unsigned atomicMax (handles negatives)
static __device__ __forceinline__ unsigned encode_f(float f) {
  unsigned u = __float_as_uint(f);
  return (u & 0x80000000u) ? ~u : (u | 0x80000000u);
}
static __device__ __forceinline__ float decode_f(unsigned e) {
  return (e & 0x80000000u) ? __uint_as_float(e & 0x7FFFFFFFu)
                           : __uint_as_float(~e);
}

// ---------------------------------------------------------------- init scratch
__global__ void k_init(long long* __restrict__ agg, unsigned* __restrict__ p_norm,
                       int M, int B) {
  int i = blockIdx.x * blockDim.x + threadIdx.x;
  if (i < M) agg[i] = 0;
  if (i < B) p_norm[i] = 0x007FFFFFu;   // encode(-inf): segment_max identity
}

// ---------------------- per-root precompute (1 wave/root); h_roots stored bf16
__global__ void k_roots(const float* __restrict__ x,
                        const float* __restrict__ w_ego_root,
                        const float* __restrict__ w_layer_v,
                        const int*   __restrict__ batch_nodes,
                        __bf16* __restrict__ hrb, float* __restrict__ nr,
                        long long* __restrict__ sv_fx, int B, int F) {
  int wave = (blockIdx.x * blockDim.x + threadIdx.x) >> 5;
  int lane = threadIdx.x & 31;
  if (wave >= B) return;
  const float* xr = x + (size_t)batch_nodes[wave] * F;
  __bf16* hr = hrb + (size_t)wave * F;
  float nr2 = 0.f, svp = 0.f;
  for (int f = lane * 4; f < F; f += 128) {
    float4 xv = *(const float4*)(xr + f);
    float4 wr = *(const float4*)(w_ego_root + f);
    float4 wv = *(const float4*)(w_layer_v + f);
    float4 hv = make_float4(xv.x * wr.x, xv.y * wr.y, xv.z * wr.z, xv.w * wr.w);
    v4bf hb;
    hb[0] = (__bf16)hv.x; hb[1] = (__bf16)hv.y;
    hb[2] = (__bf16)hv.z; hb[3] = (__bf16)hv.w;
    *(v4bf*)(hr + f) = hb;                 // bf16 copy for the WMMA B operand
    nr2 += hv.x * hv.x + hv.y * hv.y + hv.z * hv.z + hv.w * hv.w;  // exact f32
    svp += xv.x * wv.x + xv.y * wv.y + xv.z * wv.z + xv.w * wv.w;
  }
  for (int off = 16; off >= 1; off >>= 1) {
    nr2 += __shfl_xor(nr2, off);
    svp += __shfl_xor(svp, off);
  }
  if (lane == 0) {
    nr[wave] = sqrtf(nr2);
    sv_fx[wave] = (long long)llrint((double)svp * FX_SCALE);  // exact fixed point
  }
}

// -------------------------- deterministic segment_sum via i64 fixed-point atomics
__global__ void k_edges(const int* __restrict__ edge_src,
                        const int* __restrict__ edge_dst,
                        const long long* __restrict__ sv_fx,
                        long long* __restrict__ agg, int E) {
  int e = blockIdx.x * blockDim.x + threadIdx.x;
  if (e >= E) return;
  unsigned long long v = (unsigned long long)sv_fx[edge_src[e]];
  atomicAdd((unsigned long long*)&agg[edge_dst[e]], v);
}

// ------------------------------- fused scoring: 1 wave per 16 candidate rows
__global__ void __launch_bounds__(256) k_score(
    const float* __restrict__ x, const float* __restrict__ w_ego_u,
    const float* __restrict__ w_layer_u, const float* __restrict__ n_imp,
    const float* __restrict__ budgets, const int* __restrict__ u_ids,
    const int* __restrict__ batch_ptr, const __bf16* __restrict__ hrb,
    const float* __restrict__ nr, const long long* __restrict__ agg,
    float* __restrict__ p_u, unsigned* __restrict__ p_norm, int M, int F) {
  __shared__ float s_w[2048];          // [w_ego_u | w_layer_u], F <= 1024
  int tid = threadIdx.x;
  for (int i = tid; i < 2 * F; i += blockDim.x)
    s_w[i] = (i < F) ? w_ego_u[i] : w_layer_u[i - F];
  __syncthreads();

  int wave = tid >> 5;
  int lane = tid & 31;
  int tile16 = (blockIdx.x * (blockDim.x >> 5) + wave) * 16;
  if (tile16 >= M) return;             // wave-uniform; only sync is above

  int r16 = lane & 15;                 // row-in-tile (A) == column (B)
  int h   = lane >> 4;                 // half selector
  int row  = tile16 + r16;
  int rowc = row < M ? row : (M - 1);
  int u = u_ids[rowc];
  int b = batch_ptr[rowc];
  const float*  xr = x   + (size_t)u * F;
  const __bf16* hr = hrb + (size_t)b * F;

  v8f c = {0.f, 0.f, 0.f, 0.f, 0.f, 0.f, 0.f, 0.f};
  float su = 0.f, nu2 = 0.f;

  for (int k0 = 0; k0 < F; k0 += 32) {
    // ---- A fragment (hu, bf16): lane holds K = ka..ka+7 and ka+16..ka+23
    int ka = k0 + h * 8;
    float xa[16], we[16], wl[16];
    {
      float4 t;
      t = *(const float4*)(xr + ka);        xa[0]=t.x; xa[1]=t.y; xa[2]=t.z; xa[3]=t.w;
      t = *(const float4*)(xr + ka + 4);    xa[4]=t.x; xa[5]=t.y; xa[6]=t.z; xa[7]=t.w;
      t = *(const float4*)(xr + ka + 16);   xa[8]=t.x; xa[9]=t.y; xa[10]=t.z; xa[11]=t.w;
      t = *(const float4*)(xr + ka + 20);   xa[12]=t.x; xa[13]=t.y; xa[14]=t.z; xa[15]=t.w;
      t = *(const float4*)(s_w + ka);       we[0]=t.x; we[1]=t.y; we[2]=t.z; we[3]=t.w;
      t = *(const float4*)(s_w + ka + 4);   we[4]=t.x; we[5]=t.y; we[6]=t.z; we[7]=t.w;
      t = *(const float4*)(s_w + ka + 16);  we[8]=t.x; we[9]=t.y; we[10]=t.z; we[11]=t.w;
      t = *(const float4*)(s_w + ka + 20);  we[12]=t.x; we[13]=t.y; we[14]=t.z; we[15]=t.w;
      t = *(const float4*)(s_w + F + ka);      wl[0]=t.x; wl[1]=t.y; wl[2]=t.z; wl[3]=t.w;
      t = *(const float4*)(s_w + F + ka + 4);  wl[4]=t.x; wl[5]=t.y; wl[6]=t.z; wl[7]=t.w;
      t = *(const float4*)(s_w + F + ka + 16); wl[8]=t.x; wl[9]=t.y; wl[10]=t.z; wl[11]=t.w;
      t = *(const float4*)(s_w + F + ka + 20); wl[12]=t.x; wl[13]=t.y; wl[14]=t.z; wl[15]=t.w;
    }
    v16bf av;
#pragma unroll
    for (int i = 0; i < 16; ++i) {
      float huv = xa[i] * we[i];         // hu = xu * w_ego_u (f32)
      av[i] = (__bf16)huv;               // hw RTE cvt (packs 2/instr)
      nu2 += huv * huv;                  // exact fp32 ||hu||^2 partial
      su  += xa[i] * wl[i];              // exact fp32 xu . w_layer_u partial
    }
    // ---- B fragment: column j = r16 = pre-rounded bf16 h_roots row, no VALU
    int kb = k0 + h * 8;                 // lane holds K = kb..kb+7 and kb+16..kb+23
    v4bf b0 = *(const v4bf*)(hr + kb);
    v4bf b1 = *(const v4bf*)(hr + kb + 4);
    v4bf b2 = *(const v4bf*)(hr + kb + 16);
    v4bf b3 = *(const v4bf*)(hr + kb + 20);
    v16bf bv;
#pragma unroll
    for (int i = 0; i < 4; ++i) { bv[i] = b0[i]; bv[4 + i] = b1[i]; bv[8 + i] = b2[i]; bv[12 + i] = b3[i]; }
    // C[i][i] accumulates dot(hu_i, hr_i) in f32
    c = __builtin_amdgcn_wmma_f32_16x16x32_bf16(false, av, false, bv,
                                                (short)0, c, false, false);
  }

  // fold half-K partials across lane pairs
  su  += __shfl_xor(su, 16);
  nu2 += __shfl_xor(nu2, 16);

  // extract C diagonal -> lane r16. (M,N)=(i,i): i<8 -> lane i, vgpr i;
  // i>=8 -> lane i+16, vgpr i-8.
  int srcl = (r16 < 8) ? r16 : (r16 + 16);
  int vsel = r16 & 7;
  float dot = 0.f;
#pragma unroll
  for (int r = 0; r < 8; ++r) {
    float t = __shfl(c[r], srcl);
    if (vsel == r) dot = t;
  }

  if (h == 0 && row < M) {
    float nrb = nr[b];
    float nu  = sqrtf(nu2);
    float ego = dot / (fmaxf(nrb, COS_EPS) * fmaxf(nu, COS_EPS));
    float aggv  = (float)((double)agg[row] * FX_INV);
    float layer = tanhf(aggv + su);
    float p = (0.5f * ego + 0.5f * layer) * n_imp[u] * (budgets[b] * (1.0f / 200.0f));
    p_u[row] = p;
    atomicMax(&p_norm[b], encode_f(p));
  }
}

// ---------------------------------------------------------------- finalize
__global__ void k_final(const float* __restrict__ p_u,
                        const unsigned* __restrict__ p_norm,
                        const int* __restrict__ batch_ptr,
                        float* __restrict__ out, int M) {
  int m = blockIdx.x * blockDim.x + threadIdx.x;
  if (m >= M) return;
  float pn = decode_f(p_norm[batch_ptr[m]]);
  float q = p_u[m] / pn + 1.0f;
  q = isnan(q) ? 0.0f : (isinf(q) ? 1.0f : q);
  q = fminf(fmaxf(q, 1e-5f), 1.0f);
  out[m] = q;
}

extern "C" void kernel_launch(void* const* d_in, const int* in_sizes, int n_in,
                              void* d_out, int out_size, void* d_ws, size_t ws_size,
                              hipStream_t stream) {
  const float* x           = (const float*)d_in[0];
  const float* w_ego_root  = (const float*)d_in[1];
  const float* w_ego_u     = (const float*)d_in[2];
  const float* w_layer_v   = (const float*)d_in[3];
  const float* w_layer_u   = (const float*)d_in[4];
  const float* n_imp       = (const float*)d_in[5];
  const float* budgets     = (const float*)d_in[6];
  const int*   batch_nodes = (const int*)d_in[7];
  const int*   u_ids       = (const int*)d_in[8];
  const int*   batch_ptr   = (const int*)d_in[9];
  const int*   edge_src    = (const int*)d_in[10];
  const int*   edge_dst    = (const int*)d_in[11];

  const int F = in_sizes[1];     // feature dim (multiple of 32; 256 here)
  const int B = in_sizes[6];     // roots
  const int M = in_sizes[8];     // candidate pairs
  const int E = in_sizes[10];    // edges

  // workspace carve-out (8-byte arrays first for alignment)
  char* w = (char*)d_ws;
  long long* agg    = (long long*)w;  w += (size_t)M * 8;
  long long* sv_fx  = (long long*)w;  w += (size_t)B * 8;
  __bf16*    hrb    = (__bf16*)w;     w += (size_t)B * F * 2;
  float*     nr     = (float*)w;      w += (size_t)B * 4;
  float*     p_u    = (float*)w;      w += (size_t)M * 4;
  unsigned*  p_norm = (unsigned*)w;
  (void)n_in; (void)out_size; (void)ws_size;

  int mx = M > B ? M : B;
  k_init <<<(mx + 255) / 256, 256, 0, stream>>>(agg, p_norm, M, B);
  k_roots<<<(B + 7) / 8, 256, 0, stream>>>(x, w_ego_root, w_layer_v, batch_nodes,
                                           hrb, nr, sv_fx, B, F);
  k_edges<<<(E + 255) / 256, 256, 0, stream>>>(edge_src, edge_dst, sv_fx, agg, E);
  int tiles = (M + 15) / 16;
  k_score<<<(tiles + 7) / 8, 256, 0, stream>>>(x, w_ego_u, w_layer_u, n_imp, budgets,
                                               u_ids, batch_ptr, hrb, nr, agg,
                                               p_u, p_norm, M, F);
  k_final<<<(M + 255) / 256, 256, 0, stream>>>(p_u, p_norm, batch_ptr,
                                               (float*)d_out, M);
}
